// QuantizedLlamaAttention_71167608095142
// MI455X (gfx1250) — matrix operand order repaired and checked by reading, not compiled, not run on hardware
//
#include <hip/hip_runtime.h>
#include <hip/hip_bf16.h>

// ---------------------------------------------------------------------------
// QuantizedLlamaAttention for MI455X (gfx1250, wave32, WMMA)
// B=1, S=2048, H=4096, NH=32, NKV=8, HD=128, GS=128, int4 weights (zero=8)
// All GEMM/attention math on v_wmma_f32_16x16x32_f16 (f32 accumulate).
// ---------------------------------------------------------------------------

typedef __attribute__((ext_vector_type(16))) _Float16 v16h;
typedef __attribute__((ext_vector_type(8)))  _Float16 v8h;
typedef __attribute__((ext_vector_type(8)))  float    v8f;

static constexpr int S_LEN = 2048;
static constexpr int H_DIM = 4096;
static constexpr int NH    = 32;
static constexpr int NKV   = 8;
static constexpr int HD    = 128;
static constexpr int KVD   = NKV * HD;  // 1024

__device__ __forceinline__ v16h combine16(v8h lo, v8h hi) {
  v16h r;
#pragma unroll
  for (int i = 0; i < 8; i++) { r[i] = lo[i]; r[i + 8] = hi[i]; }
  return r;
}

__device__ __forceinline__ v8f wmma_f16(v16h a, v16h b, v8f c) {
  // (neg_a, A, neg_b, B, c_mod, C, reuse_a, reuse_b)
  return __builtin_amdgcn_wmma_f32_16x16x32_f16(false, a, false, b, (short)0, c,
                                                false, false);
}

// ---------------------------------------------------------------------------
// Cross-lane reductions over one DPP row (16 lanes) — pure VALU, no LDS.
// quad_perm{1,0,3,2}=0xB1 (xor1), quad_perm{2,3,0,1}=0x4E (xor2),
// row_half_mirror=0x141 (xor7), row_mirror=0x140 (xor15).
// ---------------------------------------------------------------------------
#if defined(__has_builtin)
#if __has_builtin(__builtin_amdgcn_mov_dpp)
#define ATTN_USE_DPP 1
#endif
#endif

#ifdef ATTN_USE_DPP
template <int CTRL>
__device__ __forceinline__ float dpp_movf(float x) {
  int xi = __builtin_bit_cast(int, x);
  int r = __builtin_amdgcn_mov_dpp(xi, CTRL, 0xF, 0xF, true);
  return __builtin_bit_cast(float, r);
}
__device__ __forceinline__ float row_max16(float v) {
  v = fmaxf(v, dpp_movf<0xB1>(v));
  v = fmaxf(v, dpp_movf<0x4E>(v));
  v = fmaxf(v, dpp_movf<0x141>(v));
  v = fmaxf(v, dpp_movf<0x140>(v));
  return v;
}
__device__ __forceinline__ float row_sum16(float v) {
  v += dpp_movf<0xB1>(v);
  v += dpp_movf<0x4E>(v);
  v += dpp_movf<0x141>(v);
  v += dpp_movf<0x140>(v);
  return v;
}
#else
__device__ __forceinline__ float row_max16(float v) {
#pragma unroll
  for (int off = 8; off >= 1; off >>= 1) v = fmaxf(v, __shfl_xor(v, off, 32));
  return v;
}
__device__ __forceinline__ float row_sum16(float v) {
#pragma unroll
  for (int off = 8; off >= 1; off >>= 1) v += __shfl_xor(v, off, 32);
  return v;
}
#endif

// ---------------------------------------------------------------------------
// Dequant GEMM:  Y[M,N] = X[M,K] @ ((qw[K,N] - 8) * scales[K/GS, N])
// Block tile 64x64, K-step 32, 256 threads = 8 waves (4 M-tiles x 2 N-tiles).
// Double-buffered LDS: global fetch of tile t+1 overlaps WMMAs on tile t,
// one barrier per K-step.
// ---------------------------------------------------------------------------
template <typename TIn, typename TOut>
__global__ void __launch_bounds__(256, 2)
dequant_gemm_kernel(const TIn* __restrict__ X,
                    const int* __restrict__ qw,
                    const float* __restrict__ scales,
                    TOut* __restrict__ Y,
                    int M, int N, int K) {
  constexpr int BM = 64, BN = 64, BK = 32;
  __shared__ alignas(16) _Float16 lds_a[2][BM * BK];   // [m][k] row-major
  __shared__ alignas(16) _Float16 lds_bT[2][BN * BK];  // [n][k] (transposed)

  const int n0 = blockIdx.x * BN;
  const int m0 = blockIdx.y * BM;
  const int tid = threadIdx.x;
  const int lane = tid & 31;
  const int wid = tid >> 5;
  const int waveM = wid & 3;       // 0..3 (16 rows each)
  const int waveN = wid >> 2;      // 0..1 (32 cols each)
  const int half = lane >> 4;      // lane K-group
  const int l16 = lane & 15;

  // A-load mapping: 64 rows x 32 k, 8 elems/thread
  const int am = tid >> 2;
  const int ak = (tid & 3) * 8;
  // B-load mapping: 32 k x 64 n, 8 elems/thread
  const int bk = tid >> 3;
  const int bn = (tid & 7) * 8;

  float areg[8];
  float wreg[8];

  auto global_fetch = [&](int k0) {
    const TIn* ap = X + (size_t)(m0 + am) * K + (k0 + ak);
#pragma unroll
    for (int i = 0; i < 8; i++) areg[i] = (float)ap[i];
    const int*   qp = qw + (size_t)(k0 + bk) * N + (n0 + bn);
    const float* sp = scales + (size_t)((k0 + bk) >> 7) * N + (n0 + bn);
    if (k0 + BK < K) __builtin_prefetch(qp + (size_t)BK * N, 0, 1);
#pragma unroll
    for (int i = 0; i < 8; i++) wreg[i] = (float)(qp[i] - 8) * sp[i];
  };
  auto lds_put = [&](int buf) {
    _Float16* as = &lds_a[buf][am * BK + ak];
#pragma unroll
    for (int i = 0; i < 8; i++) as[i] = (_Float16)areg[i];
#pragma unroll
    for (int i = 0; i < 8; i++) lds_bT[buf][(bn + i) * BK + bk] = (_Float16)wreg[i];
  };

  v8f acc0 = {};
  v8f acc1 = {};

  // prologue: stage tile 0
  global_fetch(0);
  lds_put(0);
  __syncthreads();

  const int T = K / BK;
  for (int t = 0; t < T; t++) {
    const int cur = t & 1;
    if (t + 1 < T) global_fetch((t + 1) * BK);

    // ---- A fragment: lane holds row m = waveM*16 + l16,
    //      K elems {16*half+e}, e = 0..15 (contiguous, matches B) ----
    const _Float16* arow = &lds_a[cur][(waveM * 16 + l16) * BK];
    v16h afrag = combine16(*(const v8h*)(arow + 16 * half),
                           *(const v8h*)(arow + 16 * half + 8));

    // ---- B fragments: lane holds col n = base + l16, K range half*16.. ----
    {
      const _Float16* brow = &lds_bT[cur][(waveN * 32 + l16) * BK + half * 16];
      v16h bfrag = combine16(*(const v8h*)brow, *(const v8h*)(brow + 8));
      acc0 = wmma_f16(afrag, bfrag, acc0);
    }
    {
      const _Float16* brow = &lds_bT[cur][(waveN * 32 + 16 + l16) * BK + half * 16];
      v16h bfrag = combine16(*(const v8h*)brow, *(const v8h*)(brow + 8));
      acc1 = wmma_f16(afrag, bfrag, acc1);
    }

    if (t + 1 < T) lds_put(cur ^ 1);
    __syncthreads();
  }

  // ---- store: C layout row = (half*8 + r), col = l16 ----
#pragma unroll
  for (int r = 0; r < 8; r++) {
    const int row = m0 + waveM * 16 + half * 8 + r;
    Y[(size_t)row * N + (n0 + waveN * 32 + l16)]      = (TOut)acc0[r];
    Y[(size_t)row * N + (n0 + waveN * 32 + 16 + l16)] = (TOut)acc1[r];
  }
}

// ---------------------------------------------------------------------------
// RoPE (in place on f16 Q [S,4096] and K [S,1024])
// ---------------------------------------------------------------------------
__global__ void rope_kernel(_Float16* __restrict__ Q, _Float16* __restrict__ Kc) {
  const size_t idx = (size_t)blockIdx.x * blockDim.x + threadIdx.x;
  const int d  = (int)(idx & 63);          // 0..63
  const int hh = (int)((idx >> 6) % 40);   // 0..31 -> Q head, 32..39 -> K head
  const int s  = (int)(idx / (64 * 40));
  if (s >= S_LEN) return;

  const float freq = (float)s * __powf(10000.0f, -(float)d / 64.0f);
  float sn, cs;
  __sincosf(freq, &sn, &cs);

  _Float16* base = (hh < NH) ? (Q + (size_t)s * H_DIM + hh * HD)
                             : (Kc + (size_t)s * KVD + (hh - NH) * HD);
  const float x1 = (float)base[d];
  const float x2 = (float)base[d + 64];
  base[d]      = (_Float16)(x1 * cs - x2 * sn);
  base[d + 64] = (_Float16)(x2 * cs + x1 * sn);
}

// ---------------------------------------------------------------------------
// Flash attention (causal, GQA rep=4). One wave = one (head, 16 q rows).
// 4 waves/block, disjoint LDS slices, __launch_bounds__(128,1) for the full
// VGPR file. K-blocks of 32 keys: all K fragments preloaded (one load
// clause), 8 S-WMMAs back-to-back; DPP-row online softmax; 8 PV-WMMAs.
// ---------------------------------------------------------------------------
__global__ void __launch_bounds__(128, 1)
attn_kernel(const _Float16* __restrict__ Q,
            const _Float16* __restrict__ Kmat,
            const _Float16* __restrict__ V,
            _Float16* __restrict__ ctx) {
  __shared__ alignas(16) _Float16 lds_v[4][32 * HD];  // per-wave V block
  __shared__ alignas(16) _Float16 lds_p[4][16 * 32];  // per-wave P tile

  const int wid  = threadIdx.x >> 5;
  const int lane = threadIdx.x & 31;
  const int half = lane >> 4;
  const int l16  = lane & 15;

  const int qt  = blockIdx.x * 4 + wid;   // q tile (16 rows)
  const int h   = blockIdx.y;             // query head
  const int kvh = h >> 2;                 // kv head (rep = 4)
  const int qrow_base = qt * 16;

  const float qscale = 0.08838834764831845f;  // 1/sqrt(128), applied post-S

  // ---- Q A-fragments for the 4 K-chunks of HD (raw vector loads) ----
  v16h qf[4];
  {
    const _Float16* qptr = Q + (size_t)(qrow_base + l16) * H_DIM + h * HD;
#pragma unroll
    for (int c = 0; c < 4; c++) {
      const _Float16* p = qptr + c * 32 + 16 * half;
      qf[c] = combine16(*(const v8h*)p, *(const v8h*)(p + 8));
    }
  }

  float m_i[8], l_i[8];
  v8f o[8];
#pragma unroll
  for (int r = 0; r < 8; r++) { m_i[r] = -__builtin_inff(); l_i[r] = 0.0f; }
#pragma unroll
  for (int j = 0; j < 8; j++) o[j] = (v8f){};

  _Float16* vbuf = lds_v[wid];
  _Float16* pbuf = lds_p[wid];

  const int qmax = qrow_base + 15;
  const int nkb  = (qmax >> 5) + 1;

  for (int kb = 0; kb < nkb; kb++) {
    const int kb0 = kb << 5;

    // ---- stage V block [32 tokens][128] into LDS (1 token row per lane) ----
    {
      const _Float16* vp = V + (size_t)(kb0 + lane) * KVD + kvh * HD;
#pragma unroll
      for (int j = 0; j < 16; j++)
        *(v8h*)(vbuf + lane * HD + j * 8) = *(const v8h*)(vp + j * 8);
      if (kb + 1 < nkb) {
        __builtin_prefetch(vp + (size_t)32 * KVD, 0, 1);
        __builtin_prefetch(Kmat + (size_t)(kb0 + 32 + lane) * KVD + kvh * HD, 0, 1);
      }
    }

    // ---- preload ALL K B-fragments (single load clause), then 8 WMMAs ----
    v16h kf0[4], kf1[4];
    {
      const _Float16* kp0 = Kmat + (size_t)(kb0 + l16) * KVD + kvh * HD + half * 16;
      const _Float16* kp1 = kp0 + (size_t)16 * KVD;
#pragma unroll
      for (int c = 0; c < 4; c++) {
        kf0[c] = combine16(*(const v8h*)(kp0 + c * 32), *(const v8h*)(kp0 + c * 32 + 8));
        kf1[c] = combine16(*(const v8h*)(kp1 + c * 32), *(const v8h*)(kp1 + c * 32 + 8));
      }
    }
    v8f st0 = {}, st1 = {};
#pragma unroll
    for (int c = 0; c < 4; c++) st0 = wmma_f16(qf[c], kf0[c], st0);
#pragma unroll
    for (int c = 0; c < 4; c++) st1 = wmma_f16(qf[c], kf1[c], st1);

    // ---- scale + causal mask + online softmax (DPP-row reductions) ----
    float alpha[8];
#pragma unroll
    for (int r = 0; r < 8; r++) {
      const int qrow = qrow_base + half * 8 + r;
      const int kcol = kb0 + l16;
      float s0 = (kcol      <= qrow) ? st0[r] * qscale : -__builtin_inff();
      float s1 = (kcol + 16 <= qrow) ? st1[r] * qscale : -__builtin_inff();
      const float mnew = fmaxf(m_i[r], row_max16(fmaxf(s0, s1)));
      alpha[r] = __expf(m_i[r] - mnew);
      m_i[r] = mnew;
      const float p0 = __expf(s0 - mnew);
      const float p1 = __expf(s1 - mnew);
      l_i[r] = l_i[r] * alpha[r] + row_sum16(p0 + p1);
      // write P (C-layout -> LDS) for relayout into an A-fragment
      pbuf[(half * 8 + r) * 32 + l16]      = (_Float16)p0;
      pbuf[(half * 8 + r) * 32 + 16 + l16] = (_Float16)p1;
    }

    // ---- rescale O by alpha ----
#pragma unroll
    for (int j = 0; j < 8; j++)
#pragma unroll
      for (int r = 0; r < 8; r++) o[j][r] *= alpha[r];

    // ---- P as A-fragment (LDS ops are in-order within a wave) ----
    const _Float16* prow = pbuf + l16 * 32;
    v16h pf = combine16(*(const v8h*)(prow + 16 * half),
                        *(const v8h*)(prow + 16 * half + 8));

    // ---- O += P @ V : 8 d-tiles ----
#pragma unroll
    for (int j = 0; j < 8; j++) {
      v16h vf;
#pragma unroll
      for (int e = 0; e < 16; e++)
        vf[e] = vbuf[(half * 16 + e) * HD + j * 16 + l16];
      o[j] = wmma_f16(pf, vf, o[j]);
    }
  }

  // ---- epilogue: normalize and write ctx [S, NH*HD] as f16 ----
#pragma unroll
  for (int r = 0; r < 8; r++) {
    const float inv = 1.0f / l_i[r];
    const int row = qrow_base + half * 8 + r;
#pragma unroll
    for (int j = 0; j < 8; j++)
      ctx[(size_t)row * H_DIM + h * HD + j * 16 + l16] = (_Float16)(o[j][r] * inv);
  }
}

// ---------------------------------------------------------------------------
// Launch
// ---------------------------------------------------------------------------
extern "C" void kernel_launch(void* const* d_in, const int* in_sizes, int n_in,
                              void* d_out, int out_size, void* d_ws, size_t ws_size,
                              hipStream_t stream) {
  (void)in_sizes; (void)n_in; (void)out_size; (void)ws_size;

  const float* hid = (const float*)d_in[0];
  const float* s_q = (const float*)d_in[1];
  const float* s_k = (const float*)d_in[2];
  const float* s_v = (const float*)d_in[3];
  const float* s_o = (const float*)d_in[4];
  const int*   qwq = (const int*)d_in[5];
  const int*   qwk = (const int*)d_in[6];
  const int*   qwv = (const int*)d_in[7];
  const int*   qwo = (const int*)d_in[8];
  float* out = (float*)d_out;

  // workspace layout (40 MB): Q f16 | K f16 | V f16 | ctx f16
  char* ws = (char*)d_ws;
  _Float16* Q   = (_Float16*)(ws);
  _Float16* Kc  = (_Float16*)(ws + (size_t)16 * 1024 * 1024);
  _Float16* V   = (_Float16*)(ws + (size_t)20 * 1024 * 1024);
  _Float16* ctx = (_Float16*)(ws + (size_t)24 * 1024 * 1024);

  const dim3 blk(256);
  // QKV projections (int4 dequant GEMM -> f16)
  dequant_gemm_kernel<float, _Float16><<<dim3(H_DIM / 64, S_LEN / 64), blk, 0, stream>>>(
      hid, qwq, s_q, Q, S_LEN, NH * HD, H_DIM);
  dequant_gemm_kernel<float, _Float16><<<dim3(KVD / 64, S_LEN / 64), blk, 0, stream>>>(
      hid, qwk, s_k, Kc, S_LEN, KVD, H_DIM);
  dequant_gemm_kernel<float, _Float16><<<dim3(KVD / 64, S_LEN / 64), blk, 0, stream>>>(
      hid, qwv, s_v, V, S_LEN, KVD, H_DIM);

  // RoPE on Q and K
  {
    const size_t total = (size_t)S_LEN * (NH + NKV) * 64;
    rope_kernel<<<dim3((unsigned)(total / 256)), dim3(256), 0, stream>>>(Q, Kc);
  }

  // causal GQA flash attention
  attn_kernel<<<dim3(S_LEN / 64, NH), dim3(128), 0, stream>>>(Q, Kc, V, ctx);

  // output projection (f16 in, f32 out)
  dequant_gemm_kernel<_Float16, float><<<dim3(H_DIM / 64, S_LEN / 64), blk, 0, stream>>>(
      ctx, qwo, s_o, out, S_LEN, H_DIM, NH * HD);
}
